// MultiHeadSelfAttention_66185446031609
// MI455X (gfx1250) — matrix operand order repaired and checked by reading, not compiled, not run on hardware
//
#include <hip/hip_runtime.h>
#include <stdint.h>

// ---------------------------------------------------------------------------
// MI455X (gfx1250) multi-head self-attention, bf16 WMMA (16x16x32, fp32 acc).
// B=2, S=2048, D=512, H=8, HD=64.  Output: [out (B,S,D) fp32][attn (B,H,S,S) fp32].
// CDNA5 paths: v_wmma_f32_16x16x32_bf16, global_load_async_to_lds_b128 (+ASYNCcnt),
// tensor_store_from_lds (TDM, TENSORcnt) for the 268MB attn writeback.
// ---------------------------------------------------------------------------

#define D_MODEL 512
#define NHEADS  8
#define HDIM    64
#define SEQ     2048
#define NBATCH  2
static constexpr float ATT_SCALE = 0.125f;  // HDIM^-0.5

typedef __attribute__((ext_vector_type(16))) __bf16 v16bf;
typedef __attribute__((ext_vector_type(8)))  float  v8f;
typedef __attribute__((ext_vector_type(4)))  unsigned int v4u;
typedef __attribute__((ext_vector_type(8)))  int v8i;
typedef __attribute__((ext_vector_type(4)))  int v4i;

#define AS1 __attribute__((address_space(1)))
#define AS3 __attribute__((address_space(3)))

#if __has_builtin(__builtin_amdgcn_global_load_async_to_lds_b128)
#define HAVE_ASYNC_LDS 1
#else
#define HAVE_ASYNC_LDS 0
#endif
#if __has_builtin(__builtin_amdgcn_tensor_store_from_lds)
#define HAVE_TDM 1
#else
#define HAVE_TDM 0
#endif

static __device__ __forceinline__ v8f wmma_bf16(v16bf a, v16bf b, v8f c) {
    return __builtin_amdgcn_wmma_f32_16x16x32_bf16(
        false, a, false, b, (short)0, c, false, false);
}

// A-fragment K index for pair j (0..7), lane-half hl (0/1).
static __device__ __forceinline__ int a_kidx(int j, int hl) {
    return (j < 4 ? 2 * j : 2 * j + 8) + hl * 8;
}

// 16B global -> LDS async copy (falls back to a synchronous copy).
static __device__ __forceinline__ void g2l_16B(const char* g, char* l) {
#if HAVE_ASYNC_LDS
    __builtin_amdgcn_global_load_async_to_lds_b128(
        (AS1 v4i*)g, (AS3 v4i*)l, 0, 0);
#else
    *(uint4*)l = *(const uint4*)g;
#endif
}
static __device__ __forceinline__ void wait_async_le1() {
#if HAVE_ASYNC_LDS
#if __has_builtin(__builtin_amdgcn_s_wait_asynccnt)
    __builtin_amdgcn_s_wait_asynccnt(1);
#else
    asm volatile("s_wait_asynccnt 1" ::: "memory");
#endif
#endif
}
static __device__ __forceinline__ void wait_async_le0() {
#if HAVE_ASYNC_LDS
#if __has_builtin(__builtin_amdgcn_s_wait_asynccnt)
    __builtin_amdgcn_s_wait_asynccnt(0);
#else
    asm volatile("s_wait_asynccnt 0" ::: "memory");
#endif
#endif
}

#if HAVE_TDM
// 1-D TDM store: 256KB contiguous LDS -> contiguous global.
// Descriptor per CDNA5 ISA: data_size=8B, tensor_dim0=tile_dim0=32768.
static __device__ __forceinline__ void tdm_store_256k(unsigned lds_off,
                                                      const void* gdst) {
    const uint64_t ga = (uint64_t)(uintptr_t)gdst;
    v4u g0;
    g0[0] = 1u;                                            // count=1
    g0[1] = lds_off;                                       // lds_addr
    g0[2] = (unsigned)ga;                                  // global_addr[31:0]
    g0[3] = ((unsigned)(ga >> 32) & 0x01FFFFFFu)
            | 0x80000000u;                                 // addr[56:32] | type=2
    v8i g1;
    g1[0] = 3 << 16;                // data_size = 8 bytes
    g1[1] = (int)0x80000000;        // tensor_dim0[15:0]=0x8000 in [31:16]
    g1[2] = 1 << 16;                // tensor_dim0[31:16]=0 ; tensor_dim1[15:0]=1
    g1[3] = (int)0x80000000;        // tensor_dim1 hi=0 ; tile_dim0=0x8000
    g1[4] = 1;                      // tile_dim1=1, tile_dim2=0
    g1[5] = 32768;                  // tensor_dim0_stride lo32
    g1[6] = 0;
    g1[7] = 0;
    const v4i z4 = {0, 0, 0, 0};
#if __clang_major__ >= 23
    const v8i z8 = {0, 0, 0, 0, 0, 0, 0, 0};
    __builtin_amdgcn_tensor_store_from_lds(g0, g1, z4, z4, z8, 0);
#else
    __builtin_amdgcn_tensor_store_from_lds(g0, g1, z4, z4, 0);
#endif
}
#endif

// ---------------------------------------------------------------------------
// Kernel 1: fused QKV projection.  y = x @ W^T + b, W in {Wq,Wk,Wv}.
// One wave: 16(M) x 64(N) tile over K=512.  bf16 out in [B,H,S,HD] layout.
// ---------------------------------------------------------------------------
__global__ __launch_bounds__(256) void mha_qkv_kernel(
    const float* __restrict__ x,
    const float* __restrict__ Wq, const float* __restrict__ Wk, const float* __restrict__ Wv,
    const float* __restrict__ bq, const float* __restrict__ bk, const float* __restrict__ bv,
    __bf16* __restrict__ qws, __bf16* __restrict__ kws, __bf16* __restrict__ vws)
{
    const int lane = threadIdx.x & 31;
    const int wave = threadIdx.x >> 5;
    const int t      = blockIdx.x * 8 + wave;   // 0..6143 wave-tiles
    const int mtile  = t / 24;
    const int nchunk = t % 24;
    const int m0 = mtile * 16;
    const int mat = nchunk >> 3;                // 0=Q 1=K 2=V
    const int e0  = (nchunk & 7) * 64;

    const float* __restrict__ Wm = (mat == 0) ? Wq : (mat == 1 ? Wk : Wv);
    const float* __restrict__ bm = (mat == 0) ? bq : (mat == 1 ? bk : bv);
    __bf16* __restrict__ ow      = (mat == 0) ? qws : (mat == 1 ? kws : vws);

    const int hl  = lane >> 4;
    const int l16 = lane & 15;
    const int row = m0 + l16;

    v8f acc[4] = {};
    for (int kc = 0; kc < 16; ++kc) {
        const int k0 = kc * 32;
        v16bf a;
        #pragma unroll
        for (int j = 0; j < 8; ++j) {
            const int kk = a_kidx(j, hl);
            const float2 xv = *(const float2*)(x + (size_t)row * D_MODEL + k0 + kk);
            a[2 * j]     = (__bf16)xv.x;
            a[2 * j + 1] = (__bf16)xv.y;
        }
        #pragma unroll
        for (int nt = 0; nt < 4; ++nt) {
            const int e = e0 + nt * 16 + l16;
            v16bf b;
            #pragma unroll
            for (int j = 0; j < 8; ++j) {
                const int kk = k0 + a_kidx(j, hl);
                const float2 wv = *(const float2*)(Wm + (size_t)e * D_MODEL + kk);
                b[2 * j]     = (__bf16)wv.x;
                b[2 * j + 1] = (__bf16)wv.y;
            }
            acc[nt] = wmma_bf16(a, b, acc[nt]);
        }
    }
    #pragma unroll
    for (int nt = 0; nt < 4; ++nt) {
        const int e = e0 + nt * 16 + l16;
        const float bias = bm[e];
        const int h  = e >> 6;
        const int hd = e & 63;
        #pragma unroll
        for (int r = 0; r < 8; ++r) {
            const int m = r + hl * 8;
            const int i = m0 + m;
            const int bb = i >> 11, s = i & 2047;
            ow[(((size_t)(bb * NHEADS + h) * SEQ + s) * HDIM) + hd] =
                (__bf16)(acc[nt][r] + bias);
        }
    }
}

// ---------------------------------------------------------------------------
// Kernel 2: pos_bias[h, q] = dot(pos_emb[q,:], Wp[h,:])
// ---------------------------------------------------------------------------
__global__ __launch_bounds__(256) void mha_posbias_kernel(
    const float* __restrict__ pos_emb, const float* __restrict__ Wp,
    float* __restrict__ pb)
{
    const int t = blockIdx.x * 256 + threadIdx.x;
    const int h = t >> 11;
    const int q = t & 2047;
    float s = 0.0f;
    #pragma unroll 4
    for (int d = 0; d < D_MODEL; d += 4) {
        const float4 pe = *(const float4*)(pos_emb + (size_t)q * D_MODEL + d);
        const float4 wp = *(const float4*)(Wp      + (size_t)h * D_MODEL + d);
        s += pe.x * wp.x + pe.y * wp.y + pe.z * wp.z + pe.w * wp.w;
    }
    pb[h * SEQ + q] = s;
}

// ---------------------------------------------------------------------------
// Kernel 3: fused attention, 32-query tile per workgroup of one (b,h).
//  P1: S = Q K^T (WMMA) -> 256KB LDS (+scale/bias/mask)
//  P2: row softmax (wave shfl reductions)
//  TDM: one tensor_store_from_lds streams P (256KB) to d_out on TENSORcnt
//  P3: ctx = P V (WMMA); V double-buffered via global_load_async_to_lds_b128
// ---------------------------------------------------------------------------
__global__ __launch_bounds__(256) void mha_attn_kernel(
    const __bf16* __restrict__ qws, const __bf16* __restrict__ kws,
    const __bf16* __restrict__ vws, const unsigned char* __restrict__ mask,
    const float* __restrict__ pb, float* __restrict__ attn_out,
    __bf16* __restrict__ aows)
{
    extern __shared__ float sc[];               // [32][2048] scores/probs (256KB)
    __bf16* vbuf = (__bf16*)(sc + 32 * SEQ);    // 2 x [32][64] bf16 (8KB)

    const int qt = blockIdx.x;                  // 0..63
    const int h  = blockIdx.y;                  // 0..7
    const int bb = blockIdx.z;                  // 0..1
    const int q0 = qt * 32;
    const int lane = threadIdx.x & 31;
    const int wave = threadIdx.x >> 5;
    const int hl  = lane >> 4;
    const int l16 = lane & 15;

    const __bf16* __restrict__ Q = qws + (size_t)(bb * NHEADS + h) * SEQ * HDIM;
    const __bf16* __restrict__ K = kws + (size_t)(bb * NHEADS + h) * SEQ * HDIM;
    const __bf16* __restrict__ V = vws + (size_t)(bb * NHEADS + h) * SEQ * HDIM;

    // ---- Phase 1: scores = Q K^T ---------------------------------------
    {
        const int mb   = wave & 1;
        const int rowq = q0 + mb * 16 + l16;
        v16bf a0, a1;
        #pragma unroll
        for (int j = 0; j < 8; ++j) {
            const int kk = a_kidx(j, hl);
            a0[2 * j]     = Q[(size_t)rowq * HDIM + kk];
            a0[2 * j + 1] = Q[(size_t)rowq * HDIM + kk + 1];
            a1[2 * j]     = Q[(size_t)rowq * HDIM + 32 + kk];
            a1[2 * j + 1] = Q[(size_t)rowq * HDIM + 32 + kk + 1];
        }
        for (int nt = (wave >> 1); nt < 128; nt += 4) {
            const int key = nt * 16 + l16;
            v16bf b0, b1;
            #pragma unroll
            for (int j = 0; j < 8; ++j) {
                const int kk = a_kidx(j, hl);
                b0[2 * j]     = K[(size_t)key * HDIM + kk];
                b0[2 * j + 1] = K[(size_t)key * HDIM + kk + 1];
                b1[2 * j]     = K[(size_t)key * HDIM + 32 + kk];
                b1[2 * j + 1] = K[(size_t)key * HDIM + 32 + kk + 1];
            }
            v8f accv = {};
            accv = wmma_bf16(a0, b0, accv);
            accv = wmma_bf16(a1, b1, accv);
            const float mk = mask[bb * SEQ + key] ? -1e30f : 0.0f;
            #pragma unroll
            for (int r = 0; r < 8; ++r) {
                const int m = mb * 16 + r + hl * 8;
                sc[m * SEQ + key] =
                    accv[r] * ATT_SCALE + pb[h * SEQ + q0 + m] + mk;
            }
        }
    }
    __syncthreads();

    // Prefetch V chunks 0,1 (async, overlaps with softmax).  A chunk is
    // 32 keys x 64 hd bf16 = 4KB = 256 threads x 16B.
    {
        const int tid = threadIdx.x;
        g2l_16B((const char*)V + (size_t)0 * 4096 + tid * 16,
                (char*)vbuf + tid * 16);
        g2l_16B((const char*)V + (size_t)1 * 4096 + tid * 16,
                (char*)vbuf + 4096 + tid * 16);
    }

    // ---- Phase 2: softmax, one wave handles 4 rows ----------------------
    for (int rr = 0; rr < 4; ++rr) {
        const int m = wave * 4 + rr;
        float mx = -1e30f;
        for (int k = lane; k < SEQ; k += 32) mx = fmaxf(mx, sc[m * SEQ + k]);
        #pragma unroll
        for (int o = 16; o; o >>= 1) mx = fmaxf(mx, __shfl_xor(mx, o, 32));
        float sum = 0.0f;
        for (int k = lane; k < SEQ; k += 32) {
            const float e = __expf(sc[m * SEQ + k] - mx);
            sc[m * SEQ + k] = e;
            sum += e;
        }
        #pragma unroll
        for (int o = 16; o; o >>= 1) sum += __shfl_xor(sum, o, 32);
        const float inv = 1.0f / sum;
        for (int k = lane; k < SEQ; k += 32) sc[m * SEQ + k] *= inv;
    }
    __syncthreads();

    // ---- ship P (256KB, contiguous) to d_out ----------------------------
    float* __restrict__ pdst =
        attn_out + ((size_t)(bb * NHEADS + h) * SEQ + q0) * SEQ;
#if HAVE_TDM
    if (wave == 0) {
        // TDM streams LDS->HBM on TENSORcnt while we compute P*V below.
        tdm_store_256k((unsigned)(size_t)(AS3 char*)sc, pdst);
    }
#else
    {
        float4* __restrict__ dst = (float4*)pdst;
        const float4* __restrict__ src = (const float4*)sc;
        for (int i = threadIdx.x; i < 32 * SEQ / 4; i += 256) dst[i] = src[i];
    }
#endif

    // ---- Phase 3: ctx = P V, V pipelined through LDS --------------------
    {
        const int mb = wave >> 2;               // 0..1
        const int n0 = (wave & 3) * 16;         // hd column base
        const int rowm = mb * 16 + l16;
        const int tid = threadIdx.x;
        v8f acc = {};
        for (int kc = 0; kc < 64; ++kc) {
            if (kc < 63) wait_async_le1(); else wait_async_le0();
            __syncthreads();                    // chunk kc resident for all waves
            const __bf16* vc = vbuf + (kc & 1) * (32 * HDIM);
            const int kb = kc * 32;
            v16bf a, b;
            #pragma unroll
            for (int j = 0; j < 8; ++j) {
                const int kk = a_kidx(j, hl);
                const float2 p = *(const float2*)(sc + rowm * SEQ + kb + kk);
                a[2 * j]     = (__bf16)p.x;
                a[2 * j + 1] = (__bf16)p.y;
                const int kl = a_kidx(j, hl);          // local key in chunk
                b[2 * j]     = vc[kl * HDIM + n0 + l16];
                b[2 * j + 1] = vc[(kl + 1) * HDIM + n0 + l16];
            }
            acc = wmma_bf16(a, b, acc);
            __syncthreads();                    // all waves done with chunk kc
            if (kc + 2 < 64) {                  // refill the buffer just freed
                g2l_16B((const char*)V + (size_t)(kc + 2) * 4096 + tid * 16,
                        (char*)vbuf + (kc & 1) * 4096 + tid * 16);
            }
        }
        #pragma unroll
        for (int r = 0; r < 8; ++r) {
            const int m = mb * 16 + r + hl * 8;
            aows[((size_t)bb * SEQ + q0 + m) * D_MODEL + h * HDIM + n0 + l16] =
                (__bf16)acc[r];
        }
    }
    // S_ENDPGM performs an implicit wait-idle (TENSORcnt/ASYNCcnt drain).
}

// ---------------------------------------------------------------------------
// Kernel 4: output projection  out = ctx @ Wo^T + bo (fp32)
// ---------------------------------------------------------------------------
__global__ __launch_bounds__(256) void mha_oproj_kernel(
    const __bf16* __restrict__ aows, const float* __restrict__ Wo,
    const float* __restrict__ bo, float* __restrict__ out)
{
    const int lane = threadIdx.x & 31;
    const int wave = threadIdx.x >> 5;
    const int t      = blockIdx.x * 8 + wave;
    const int mtile  = t >> 3;
    const int nchunk = t & 7;
    const int m0 = mtile * 16;
    const int e0 = nchunk * 64;
    const int hl  = lane >> 4;
    const int l16 = lane & 15;
    const int row = m0 + l16;

    v8f acc[4] = {};
    for (int kc = 0; kc < 16; ++kc) {
        const int k0 = kc * 32;
        v16bf a;
        #pragma unroll
        for (int j = 0; j < 8; ++j) {
            const int kk = k0 + a_kidx(j, hl);
            a[2 * j]     = aows[(size_t)row * D_MODEL + kk];
            a[2 * j + 1] = aows[(size_t)row * D_MODEL + kk + 1];
        }
        #pragma unroll
        for (int nt = 0; nt < 4; ++nt) {
            const int e = e0 + nt * 16 + l16;
            v16bf b;
            #pragma unroll
            for (int j = 0; j < 8; ++j) {
                const int kk = k0 + a_kidx(j, hl);
                const float2 wv = *(const float2*)(Wo + (size_t)e * D_MODEL + kk);
                b[2 * j]     = (__bf16)wv.x;
                b[2 * j + 1] = (__bf16)wv.y;
            }
            acc[nt] = wmma_bf16(a, b, acc[nt]);
        }
    }
    #pragma unroll
    for (int nt = 0; nt < 4; ++nt) {
        const int e = e0 + nt * 16 + l16;
        const float bias = bo[e];
        #pragma unroll
        for (int r = 0; r < 8; ++r) {
            const int m = r + hl * 8;
            out[(size_t)(m0 + m) * D_MODEL + e] = acc[nt][r] + bias;
        }
    }
}

// ---------------------------------------------------------------------------
extern "C" void kernel_launch(void* const* d_in, const int* in_sizes, int n_in,
                              void* d_out, int out_size, void* d_ws, size_t ws_size,
                              hipStream_t stream) {
    const float* x   = (const float*)d_in[0];
    const float* pos = (const float*)d_in[1];
    const unsigned char* mask = (const unsigned char*)d_in[2];
    const float* Wq = (const float*)d_in[3];
    const float* bq = (const float*)d_in[4];
    const float* Wk = (const float*)d_in[5];
    const float* bk = (const float*)d_in[6];
    const float* Wv = (const float*)d_in[7];
    const float* bv = (const float*)d_in[8];
    const float* Wo = (const float*)d_in[9];
    const float* bo = (const float*)d_in[10];
    const float* Wp = (const float*)d_in[11];

    char* ws = (char*)d_ws;
    __bf16* qws  = (__bf16*)(ws);                    // [B,H,S,HD] bf16 : 4 MiB
    __bf16* kws  = (__bf16*)(ws + (4u << 20));
    __bf16* vws  = (__bf16*)(ws + (8u << 20));
    __bf16* aows = (__bf16*)(ws + (12u << 20));      // [B*S, D] bf16 : 4 MiB
    float*  pb   = (float*)(ws + (16u << 20));       // [H,S] f32 : 64 KiB

    float* out  = (float*)d_out;
    float* attn = out + (size_t)NBATCH * SEQ * D_MODEL;

    const size_t att_lds = (size_t)32 * SEQ * sizeof(float)   // scores
                         + (size_t)2 * 32 * HDIM * 2;         // V double buffer

    mha_qkv_kernel<<<768, 256, 0, stream>>>(x, Wq, Wk, Wv, bq, bk, bv,
                                            qws, kws, vws);
    mha_posbias_kernel<<<64, 256, 0, stream>>>(pos, Wp, pb);
    mha_attn_kernel<<<dim3(64, NHEADS, NBATCH), 256, att_lds, stream>>>(
        qws, kws, vws, mask, pb, attn, aows);
    mha_oproj_kernel<<<256, 256, 0, stream>>>(aows, Wo, bo, out);
}
